// FeatureAttentionLayer_46935402610805
// MI455X (gfx1250) — matrix operand order repaired and checked by reading, not compiled, not run on hardware
//
#include <hip/hip_runtime.h>
#include <math.h>

#define BB 8
#define WW 128
#define KK 2048
#define DD 256
#define MT 4                 // i-tiles of 16 per block => 64 output rows per block

typedef __bf16 bf16_t;
typedef __attribute__((ext_vector_type(16))) __bf16 v16bf;
typedef __attribute__((ext_vector_type(8)))  __bf16 v8bf;
typedef __attribute__((ext_vector_type(8)))  float  v8f;

// ---------------- wave reductions (wave32) ----------------
__device__ __forceinline__ float waveMax(float v) {
#pragma unroll
    for (int o = 16; o > 0; o >>= 1) v = fmaxf(v, __shfl_xor(v, o, 32));
    return v;
}
__device__ __forceinline__ float waveSum(float v) {
#pragma unroll
    for (int o = 16; o > 0; o >>= 1) v += __shfl_xor(v, o, 32);
    return v;
}

// ---------------- kernel 1: w2a[w] = sum_d W_lin[W+w][d] * a[d] ----------------
__global__ __launch_bounds__(256) void k_w2a(const float* __restrict__ Wlin,
                                             const float* __restrict__ a,
                                             float* __restrict__ w2a) {
    __shared__ float as[DD];
    int t = threadIdx.x;
    as[t] = a[t];
    __syncthreads();
    if (t < WW) {
        const float* row = Wlin + (size_t)(WW + t) * DD;
        float acc = 0.f;
#pragma unroll 8
        for (int d = 0; d < DD; ++d) acc += row[d] * as[d];
        w2a[t] = acc;
    }
}

// ---------------- kernel 2: u2[b][k] = sum_w x[b][w][k] * w2a[w] ----------------
__global__ __launch_bounds__(256) void k_u2(const float* __restrict__ x,
                                            const float* __restrict__ w2a,
                                            float* __restrict__ u2) {
    __shared__ float ws[WW];
    int tid = threadIdx.x;
    if (tid < WW) ws[tid] = w2a[tid];
    __syncthreads();
    int b = blockIdx.y;
    int k = blockIdx.x * 256 + tid;
    const float* xb = x + (size_t)b * WW * KK + k;
    float acc = 0.f;
#pragma unroll 8
    for (int w = 0; w < WW; ++w) acc += xb[(size_t)w * KK] * ws[w];
    u2[b * KK + k] = acc;
}

// ---------------- kernel 3: softmax row stats m[b][i], l[b][i] ----------------
__global__ __launch_bounds__(256) void k_stats(const float* __restrict__ bias,
                                               const float* __restrict__ u2,
                                               float* __restrict__ m,
                                               float* __restrict__ l) {
    const int i = blockIdx.x;
    const int tid = threadIdx.x;
    const int lane = tid & 31, wave = tid >> 5;
    float br[8];
    const float* brow = bias + (size_t)i * KK;
#pragma unroll
    for (int s = 0; s < 8; ++s) br[s] = brow[tid + s * 256];
    __shared__ float red[8];
    for (int b = 0; b < BB; ++b) {
        const float* ub = u2 + b * KK;
        float t[8];
        float vmax = -3.0e38f;
#pragma unroll
        for (int s = 0; s < 8; ++s) {
            t[s] = ub[tid + s * 256] + br[s];
            vmax = fmaxf(vmax, t[s]);
        }
        vmax = waveMax(vmax);
        if (lane == 0) red[wave] = vmax;
        __syncthreads();
        float bm = red[0];
#pragma unroll
        for (int q = 1; q < 8; ++q) bm = fmaxf(bm, red[q]);
        __syncthreads();
        float s = 0.f;
#pragma unroll
        for (int q = 0; q < 8; ++q) s += __expf(t[q] - bm);
        s = waveSum(s);
        if (lane == 0) red[wave] = s;
        __syncthreads();
        float bs = red[0] + red[1] + red[2] + red[3] + red[4] + red[5] + red[6] + red[7];
        __syncthreads();
        if (tid == 0) {
            m[b * KK + i] = bm;
            l[b * KK + i] = bs;
        }
    }
}

// ---------------- kernel 4: fused softmax-recompute + bf16 WMMA GEMM ----------------
// Block: 8 waves. 64 output rows (4 x 16 i-tiles); wave w -> output cols [16w,16w+16).
// Double-buffered A stage (one barrier per j-step); all A-fragment LDS loads are
// issued before the WMMA drain so dscnt waits are partial, not full.

__device__ __forceinline__ v16bf loadB16(const float* __restrict__ p) {
    const float4* q = (const float4*)p;
    float4 a0 = q[0], a1 = q[1], a2 = q[2], a3 = q[3];
    v16bf r;
    r[0] = (bf16_t)a0.x;  r[1] = (bf16_t)a0.y;  r[2] = (bf16_t)a0.z;  r[3] = (bf16_t)a0.w;
    r[4] = (bf16_t)a1.x;  r[5] = (bf16_t)a1.y;  r[6] = (bf16_t)a1.z;  r[7] = (bf16_t)a1.w;
    r[8] = (bf16_t)a2.x;  r[9] = (bf16_t)a2.y;  r[10] = (bf16_t)a2.z; r[11] = (bf16_t)a2.w;
    r[12] = (bf16_t)a3.x; r[13] = (bf16_t)a3.y; r[14] = (bf16_t)a3.z; r[15] = (bf16_t)a3.w;
    return r;
}

__device__ __forceinline__ v16bf loadA16(const bf16_t* __restrict__ rowBase, int g, int koff) {
    // A layout (contiguous, matches B): lane half g -> K = koff+g*16+[0..15]
    union {
        v16bf v;
        struct { v8bf lo, hi; } p;
    } u;
    u.p.lo = *(const v8bf*)(rowBase + koff + g * 16);
    u.p.hi = *(const v8bf*)(rowBase + koff + g * 16 + 8);
    return u.v;
}

__global__ __launch_bounds__(256) void attn_gemm(const float* __restrict__ x,
                                                 const float* __restrict__ bias,
                                                 const float* __restrict__ u2,
                                                 const float* __restrict__ mrow,
                                                 const float* __restrict__ lrow,
                                                 float* __restrict__ out) {
    const int i0 = blockIdx.x * (16 * MT);
    const int b = blockIdx.y;
    const int tid = threadIdx.x;
    const int lane = tid & 31;
    const int wave = tid >> 5;
    const int g = lane >> 4;   // K-half group
    const int n = lane & 15;   // B column / A row for this lane
    const int w0 = wave * 16;

    __shared__ __align__(16) bf16_t Ab[2][64 * 72];  // 2 x 9216 B, row stride 144 B
    __shared__ __align__(16) float  CST[128 * 69];   // 35328 B, stride 69 (conflict-free)

    const float* xb = x + (size_t)b * WW * KK;
    const float* u2b = u2 + b * KK;

    // A-fill geometry: thread covers rows {frow, frow+16, frow+32, frow+48}, cols [fcol,fcol+4)
    const int frow = tid >> 4;        // 0..15
    const int fcol = (tid & 15) * 4;  // 0..60
    float mr[MT];
#pragma unroll
    for (int rr = 0; rr < MT; ++rr) mr[rr] = mrow[b * KK + i0 + frow + 16 * rr];

    auto fillA = [&](int buf, int j) {
        const float* ur = u2b + j + fcol;
        float u0 = ur[0], u1 = ur[1], u2v = ur[2], u3 = ur[3];
#pragma unroll
        for (int rr = 0; rr < MT; ++rr) {
            const int row = frow + 16 * rr;
            const float* br = bias + (size_t)(i0 + row) * KK + j + fcol;
            bf16_t* dst = &Ab[buf][row * 72 + fcol];
            dst[0] = (bf16_t)__expf(u0 + br[0] - mr[rr]);
            dst[1] = (bf16_t)__expf(u1 + br[1] - mr[rr]);
            dst[2] = (bf16_t)__expf(u2v + br[2] - mr[rr]);
            dst[3] = (bf16_t)__expf(u3 + br[3] - mr[rr]);
        }
    };

    v8f acc[MT];
#pragma unroll
    for (int t = 0; t < MT; ++t) acc[t] = (v8f){0.f, 0.f, 0.f, 0.f, 0.f, 0.f, 0.f, 0.f};

    fillA(0, 0);  // prologue

    const int NIT = KK / 64;
    for (int it = 0; it < NIT; ++it) {
        const int j = it * 64;
        const int p = it & 1;
        __syncthreads();  // buffer p fully staged; previous reads of buffer p^1 done

        const float* bp = xb + (size_t)(w0 + n) * KK + j + g * 16;
        v16bf B0 = loadB16(bp);
        v16bf B1 = loadB16(bp + 32);
        __builtin_prefetch(bp + 64, 0, 3);  // next j-tile of x

        // issue ALL A-fragment LDS loads first -> partial dscnt waits during drain
        v16bf A0v[MT], A1v[MT];
#pragma unroll
        for (int t = 0; t < MT; ++t) {
            const bf16_t* aRow = &Ab[p][(t * 16 + n) * 72];
            A0v[t] = loadA16(aRow, g, 0);
            A1v[t] = loadA16(aRow, g, 32);
        }
#pragma unroll
        for (int t = 0; t < MT; ++t) {
            acc[t] = __builtin_amdgcn_wmma_f32_16x16x32_bf16(false, A0v[t], false, B0,
                                                             (short)0, acc[t], false, false);
            acc[t] = __builtin_amdgcn_wmma_f32_16x16x32_bf16(false, A1v[t], false, B1,
                                                             (short)0, acc[t], false, false);
        }

        // overlap: stage next tile into the other buffer while WMMAs drain
        if (it + 1 < NIT) fillA(p ^ 1, j + 64);
    }

    // stage C through LDS for k-contiguous coalesced stores (output is [B][W][K])
#pragma unroll
    for (int t = 0; t < MT; ++t) {
#pragma unroll
        for (int r = 0; r < 8; ++r) {
            // C layout: VGPR r -> M = r + 8*g (i'), N = n (w)
            CST[(w0 + n) * 69 + t * 16 + r + 8 * g] = acc[t][r];
        }
    }
    __syncthreads();

    {
        const int w = tid >> 1;
        const int half = tid & 1;
        const float* cs = CST + w * 69 + half * 32;
        const float* lr = lrow + b * KK + i0 + half * 32;
        float* op = out + (size_t)b * WW * KK + (size_t)w * KK + i0 + half * 32;
#pragma unroll
        for (int blk = 0; blk < 8; ++blk) {
            float v[4];
#pragma unroll
            for (int q = 0; q < 4; ++q) {
                const int idx = blk * 4 + q;
                float hv = cs[idx] * __builtin_amdgcn_rcpf(lr[idx]);   // softmax normalize
                v[q] = __builtin_amdgcn_rcpf(1.0f + __expf(-hv));      // sigmoid
            }
            *(float4*)(op + blk * 4) = make_float4(v[0], v[1], v[2], v[3]);
        }
    }
}

// ---------------- launcher ----------------
extern "C" void kernel_launch(void* const* d_in, const int* in_sizes, int n_in,
                              void* d_out, int out_size, void* d_ws, size_t ws_size,
                              hipStream_t stream) {
    const float* x    = (const float*)d_in[0];  // (B, W, K)
    const float* Wlin = (const float*)d_in[1];  // (2W, D)
    // d_in[2] = b_lin: provably irrelevant (softmax shift invariance along reduced axis)
    const float* a    = (const float*)d_in[3];  // (D, 1)
    const float* bias = (const float*)d_in[4];  // (K, K)
    float* out = (float*)d_out;                 // (B, W, K)

    float* ws  = (float*)d_ws;
    float* w2a = ws;                        // 128
    float* u2  = ws + 128;                  // B*K = 16384
    float* m   = ws + 128 + 16384;          // B*K
    float* l   = ws + 128 + 2 * 16384;      // B*K   (total ~193 KB)

    k_w2a<<<1, 256, 0, stream>>>(Wlin, a, w2a);
    k_u2<<<dim3(KK / 256, BB), 256, 0, stream>>>(x, w2a, u2);
    k_stats<<<KK, 256, 0, stream>>>(bias, u2, m, l);
    attn_gemm<<<dim3(KK / (16 * MT), BB), 256, 0, stream>>>(x, bias, u2, m, l, out);
}
